// FlexMoERouter_24215025615344
// MI455X (gfx1250) — matrix operand order repaired and checked by reading, not compiled, not run on hardware
//
#include <hip/hip_runtime.h>
#include <hip/hip_bf16.h>

// FlexMoE router for MI455X (gfx1250, wave32).
#define T_TOK   16384      // B*S
#define H_DIM   2048
#define HM_DIM  4096       // H*M
#define M_MOD   2
#define E_EXP   16
#define LN_EPS  1e-5f

typedef __attribute__((ext_vector_type(16))) __bf16 bf16x16;
typedef __attribute__((ext_vector_type(8)))  float  f32x8;
typedef __attribute__((ext_vector_type(4)))  int    i32x4;

union Frag32B { bf16x16 v; uint4 q[2]; };

__device__ __forceinline__ unsigned short f2bf(float f) {
  unsigned int u = __float_as_uint(f);
  u += 0x7FFFu + ((u >> 16) & 1u);          // round-to-nearest-even
  return (unsigned short)(u >> 16);
}

// CDNA5 async global->LDS path (ASYNCcnt), with safe fallback.
#if __has_builtin(__builtin_amdgcn_global_load_async_to_lds_b128)
#define ASYNC_LDS 1
#else
#define ASYNC_LDS 0
#endif

__device__ __forceinline__ void wait_async0() {
#if __has_builtin(__builtin_amdgcn_s_wait_asynccnt)
  __builtin_amdgcn_s_wait_asynccnt(0);
#else
  asm volatile("s_wait_asynccnt 0x0" ::: "memory");
#endif
}

// ---------------------------------------------------------------- utilities
__global__ void zero_kernel(float* __restrict__ p, int n) {
  int i = blockIdx.x * blockDim.x + threadIdx.x;
  if (i < n) p[i] = 0.0f;
}

__global__ void cvt_f32_bf16(const float* __restrict__ src,
                             unsigned short* __restrict__ dst, int n) {
  int i = blockIdx.x * blockDim.x + threadIdx.x;
  int stride = gridDim.x * blockDim.x;
  for (; i < n; i += stride) dst[i] = f2bf(src[i]);
}

// ------------------------------------------------- fusion MLP partial sums
// Block tile: 128 tokens x 128 cols, K through double-buffered LDS in
// 64-wide chunks staged with async global->LDS copies. 8 waves = 2(M)x4(N);
// per-wave register tile 64x32 = acc[4][2].
#define FUS_THREADS 256
#define MBLK 128
#define NBLK 128
#define KC   64
#define LDA  (KC + 8)      // 72 elems = 144 B row stride: 16B aligned, banks spread

__device__ __forceinline__ void stage_fusion_chunk(
    const unsigned short* __restrict__ mi_bf,
    const unsigned short* __restrict__ fw1,
    unsigned short* la, unsigned short* lb,
    int tid, int tok0, int n0, int k0)
{
  #pragma unroll
  for (int j = 0; j < 4; ++j) {
    int q = tid + FUS_THREADS * j;                       // uint4 index
    int row = q >> 3, qc = q & 7;
    const unsigned short* ga = mi_bf + (size_t)(tok0 + row) * HM_DIM + k0 + qc * 8;
    const unsigned short* gb = fw1   + (size_t)(n0  + row) * HM_DIM + k0 + qc * 8;
    unsigned short* da = la + row * LDA + qc * 8;
    unsigned short* db = lb + row * LDA + qc * 8;
#if ASYNC_LDS
    __builtin_amdgcn_global_load_async_to_lds_b128(
        (__attribute__((address_space(1))) i32x4*)(void*)ga,
        (__attribute__((address_space(3))) i32x4*)(void*)da, 0, 0);
    __builtin_amdgcn_global_load_async_to_lds_b128(
        (__attribute__((address_space(1))) i32x4*)(void*)gb,
        (__attribute__((address_space(3))) i32x4*)(void*)db, 0, 0);
#else
    *(uint4*)da = *(const uint4*)ga;
    *(uint4*)db = *(const uint4*)gb;
#endif
  }
}

__global__ void __launch_bounds__(FUS_THREADS)
fusion_kernel(const unsigned short* __restrict__ mi_bf,  // [T, 4096] bf16
              const unsigned short* __restrict__ fw1,    // [2048, 4096] bf16
              const float* __restrict__ fb1,             // [2048]
              const float* __restrict__ fw2,             // [2, 2048]
              float* __restrict__ mwacc)                 // [T, 2] atomic acc
{
  __shared__ unsigned short lds_a[2][MBLK * LDA];        // 36.9 KB
  __shared__ unsigned short lds_b[2][NBLK * LDA];        // 36.9 KB
  __shared__ float lds_red[4][MBLK][2];                  // 4 KB

  const int tid  = threadIdx.x;
  const int wave = tid >> 5;
  const int lane = tid & 31;
  const int half = lane >> 4;
  const int l16  = lane & 15;
  const int wm   = wave >> 2;                            // 0..1 (M group)
  const int wn   = wave & 3;                             // 0..3 (N group)
  const int bn   = blockIdx.x & 15;                      // N slice
  const int bm   = blockIdx.x >> 4;                      // M block
  const int tok0 = bm * MBLK;
  const int n0   = bn * NBLK;

  f32x8 acc[4][2];
  #pragma unroll
  for (int a = 0; a < 4; ++a)
    #pragma unroll
    for (int c = 0; c < 2; ++c)
      #pragma unroll
      for (int j = 0; j < 8; ++j) acc[a][c][j] = 0.0f;

  // prologue: stage chunk 0 into buffer 0
  stage_fusion_chunk(mi_bf, fw1, lds_a[0], lds_b[0], tid, tok0, n0, 0);

  const int NCHUNK = HM_DIM / KC;
  for (int kc = 0; kc < NCHUNK; ++kc) {
    const int cur = kc & 1;
#if ASYNC_LDS
    wait_async0();                 // my async copies into buf 'cur' landed
#endif
    __syncthreads();               // everyone's copies visible

    if (kc + 1 < NCHUNK)           // prefetch next chunk into the other buffer
      stage_fusion_chunk(mi_bf, fw1, lds_a[cur ^ 1], lds_b[cur ^ 1],
                         tid, tok0, n0, (kc + 1) * KC);

    const unsigned short* la = lds_a[cur];
    const unsigned short* lb = lds_b[cur];
    #pragma unroll
    for (int ks = 0; ks < KC / 32; ++ks) {
      const int k = ks * 32;
      Frag32B af[4];
      #pragma unroll
      for (int a = 0; a < 4; ++a) {
        const unsigned short* ap = la + (wm * 64 + a * 16 + l16) * LDA + k + half * 8;
        af[a].q[0] = *(const uint4*)(ap);
        af[a].q[1] = *(const uint4*)(ap + 16);
      }
      #pragma unroll
      for (int c = 0; c < 2; ++c) {
        Frag32B bf;
        const unsigned short* bp = lb + (wn * 32 + c * 16 + l16) * LDA + k + half * 8;
        bf.q[0] = *(const uint4*)(bp);
        bf.q[1] = *(const uint4*)(bp + 16);
        #pragma unroll
        for (int a = 0; a < 4; ++a)
          acc[a][c] = __builtin_amdgcn_wmma_f32_16x16x32_bf16(
              false, af[a].v, false, bf.v, (short)0, acc[a][c], false, false);
      }
    }
  }

  // bias + ReLU + fold h . f_w2 over this block's 128 columns
  float mw0[4][8], mw1[4][8];
  #pragma unroll
  for (int a = 0; a < 4; ++a)
    #pragma unroll
    for (int r = 0; r < 8; ++r) { mw0[a][r] = 0.0f; mw1[a][r] = 0.0f; }
  #pragma unroll
  for (int c = 0; c < 2; ++c) {
    const int n = n0 + wn * 32 + c * 16 + l16;
    const float bias = fb1[n];
    const float w20  = fw2[n];
    const float w21  = fw2[H_DIM + n];
    #pragma unroll
    for (int a = 0; a < 4; ++a)
      #pragma unroll
      for (int r = 0; r < 8; ++r) {
        float h = acc[a][c][r] + bias;
        h = h > 0.0f ? h : 0.0f;
        mw0[a][r] = fmaf(h, w20, mw0[a][r]);
        mw1[a][r] = fmaf(h, w21, mw1[a][r]);
      }
  }
  #pragma unroll
  for (int off = 1; off < 16; off <<= 1)
    #pragma unroll
    for (int a = 0; a < 4; ++a)
      #pragma unroll
      for (int r = 0; r < 8; ++r) {
        mw0[a][r] += __shfl_xor(mw0[a][r], off, 32);
        mw1[a][r] += __shfl_xor(mw1[a][r], off, 32);
      }
  if (l16 == 0) {
    #pragma unroll
    for (int a = 0; a < 4; ++a)
      #pragma unroll
      for (int r = 0; r < 8; ++r) {
        const int row = wm * 64 + a * 16 + half * 8 + r;
        lds_red[wn][row][0] = mw0[a][r];
        lds_red[wn][row][1] = mw1[a][r];
      }
  }
  __syncthreads();
  {
    const int row = tid >> 1, m = tid & 1;
    float s = lds_red[0][row][m] + lds_red[1][row][m] +
              lds_red[2][row][m] + lds_red[3][row][m];
    atomicAdd(&mwacc[(tok0 + row) * 2 + m], s);
  }
}

// ----------------------------------- modality softmax + balance-loss sums
__global__ void __launch_bounds__(256)
modweights_kernel(const float* __restrict__ mwacc, const float* __restrict__ fb2,
                  float* __restrict__ mw_out, float* __restrict__ msum)
{
  __shared__ float red0[256], red1[256];
  const int t = blockIdx.x * 256 + threadIdx.x;
  const float s0 = mwacc[t * 2 + 0] + fb2[0];
  const float s1 = mwacc[t * 2 + 1] + fb2[1];
  const float mx = fmaxf(s0, s1);
  const float e0 = __expf(s0 - mx), e1 = __expf(s1 - mx);
  const float inv = 1.0f / (e0 + e1);
  const float w0 = e0 * inv, w1 = e1 * inv;
  mw_out[t * 2 + 0] = w0;
  mw_out[t * 2 + 1] = w1;
  red0[threadIdx.x] = w0; red1[threadIdx.x] = w1;
  __syncthreads();
  for (int st = 128; st > 0; st >>= 1) {
    if (threadIdx.x < st) {
      red0[threadIdx.x] += red0[threadIdx.x + st];
      red1[threadIdx.x] += red1[threadIdx.x + st];
    }
    __syncthreads();
  }
  if (threadIdx.x == 0) {
    atomicAdd(&msum[0], red0[0]);
    atomicAdd(&msum[1], red1[0]);
  }
}

// ----------------------------------------------------------- LayerNorm (bf16)
__global__ void __launch_bounds__(256)
ln_kernel(const float* __restrict__ x, const float* __restrict__ g,
          const float* __restrict__ b, unsigned short* __restrict__ xn)
{
  __shared__ float red[256];
  const int tok = blockIdx.x;
  const int tid = threadIdx.x;
  const float* xr = x + (size_t)tok * H_DIM;
  float lx[8];
  float s = 0.0f;
  #pragma unroll
  for (int j = 0; j < 8; ++j) { lx[j] = xr[tid + 256 * j]; s += lx[j]; }
  red[tid] = s; __syncthreads();
  for (int st = 128; st > 0; st >>= 1) {
    if (tid < st) red[tid] += red[tid + st];
    __syncthreads();
  }
  const float mu = red[0] * (1.0f / H_DIM);
  __syncthreads();
  float s2 = 0.0f;
  #pragma unroll
  for (int j = 0; j < 8; ++j) { float d = lx[j] - mu; s2 += d * d; }
  red[tid] = s2; __syncthreads();
  for (int st = 128; st > 0; st >>= 1) {
    if (tid < st) red[tid] += red[tid + st];
    __syncthreads();
  }
  const float rs = rsqrtf(red[0] * (1.0f / H_DIM) + LN_EPS);
  #pragma unroll
  for (int j = 0; j < 8; ++j) {
    const int i = tid + 256 * j;
    xn[(size_t)tok * H_DIM + i] = f2bf((lx[j] - mu) * rs * g[i] + b[i]);
  }
}

// ---------------------------- router GEMM + softmax + top-2 + expert p sums
__global__ void __launch_bounds__(256)
router_kernel(const unsigned short* __restrict__ xn,    // [T, 2048] bf16
              const unsigned short* __restrict__ mrw,   // [16, 2048] bf16
              const float* __restrict__ mrb,            // [16]
              const float* __restrict__ mw,             // [T, 2]
              float* __restrict__ out,                  // idx[T*2] p[T*2] aux
              float* __restrict__ pexp)                 // [16]
{
  __shared__ float lds_p[8][16][16];                    // 8 KB
  __shared__ float lds_pe[16];
  const int tid  = threadIdx.x;
  const int wave = tid >> 5;
  const int lane = tid & 31;
  const int half = lane >> 4;
  const int l16  = lane & 15;
  const int tile = blockIdx.x * 8 + wave;
  const int tb   = tile * 16;

  if (tid < 16) lds_pe[tid] = 0.0f;
  __syncthreads();

  f32x8 acc;
  #pragma unroll
  for (int j = 0; j < 8; ++j) acc[j] = 0.0f;

  const unsigned short* ap = xn  + (size_t)(tb + l16) * H_DIM + half * 8;
  const unsigned short* bp = mrw + (size_t)l16 * H_DIM + half * 8;
  for (int k = 0; k < H_DIM; k += 32) {
    Frag32B a, b;
    a.q[0] = *(const uint4*)(ap + k);
    a.q[1] = *(const uint4*)(ap + k + 16);
    b.q[0] = *(const uint4*)(bp + k);
    b.q[1] = *(const uint4*)(bp + k + 16);
    acc = __builtin_amdgcn_wmma_f32_16x16x32_bf16(
        false, a.v, false, b.v, (short)0, acc, false, false);
  }

  const int e    = l16;
  const float bias = mrb[e];
  const int msel = e >> 3;
  float p[8];
  #pragma unroll
  for (int r = 0; r < 8; ++r) {
    const int tok = tb + half * 8 + r;
    p[r] = (acc[r] + bias) * mw[tok * 2 + msel];
  }
  #pragma unroll
  for (int r = 0; r < 8; ++r) {
    float mx = p[r];
    for (int off = 1; off < 16; off <<= 1) mx = fmaxf(mx, __shfl_xor(mx, off, 32));
    float ex = __expf(p[r] - mx);
    float sm = ex;
    for (int off = 1; off < 16; off <<= 1) sm += __shfl_xor(sm, off, 32);
    p[r] = ex / sm;
  }
  float ls = 0.0f;
  #pragma unroll
  for (int r = 0; r < 8; ++r) {
    lds_p[wave][half * 8 + r][e] = p[r];
    ls += p[r];
  }
  atomicAdd(&lds_pe[e], ls);                            // LDS-level reduce
  __syncthreads();
  if (tid < 16) atomicAdd(&pexp[tid], lds_pe[tid]);     // 16 global atomics/blk

  if (lane < 16) {                                      // top-2 per token
    const float* pr = &lds_p[wave][lane][0];
    float b1 = -1.0f, b2 = -1.0f; int i1 = 0, i2 = 0;
    for (int j = 0; j < E_EXP; ++j) {
      const float v = pr[j];
      if (v > b1)      { b2 = b1; i2 = i1; b1 = v; i1 = j; }
      else if (v > b2) { b2 = v; i2 = j; }
    }
    const float inv = 1.0f / (b1 + b2);
    const int token = tb + lane;
    out[token * 2 + 0] = (float)i1;
    out[token * 2 + 1] = (float)i2;
    out[2 * T_TOK + token * 2 + 0] = b1 * inv;
    out[2 * T_TOK + token * 2 + 1] = b2 * inv;
  }
}

// ------------------------------------------------------------- aux loss
__global__ void finalize_kernel(const float* __restrict__ pexp,
                                const float* __restrict__ msum,
                                float* __restrict__ out)
{
  if (threadIdx.x == 0 && blockIdx.x == 0) {
    const float invT = 1.0f / (float)T_TOK;
    float ll = 0.0f;
    for (int e = 0; e < E_EXP; ++e) {
      const float pe = pexp[e] * invT;
      ll += pe * __logf(pe * (float)E_EXP + 1e-9f);
    }
    float ml = 0.0f;
    for (int m = 0; m < M_MOD; ++m) {
      const float mb = msum[m] * invT;
      ml += mb * __logf(mb * (float)M_MOD + 1e-9f);
    }
    out[2 * T_TOK * 2] = ll + 0.1f * ml;
  }
}

// ---------------------------------------------------------------- launcher
extern "C" void kernel_launch(void* const* d_in, const int* in_sizes, int n_in,
                              void* d_out, int out_size, void* d_ws, size_t ws_size,
                              hipStream_t stream) {
  const float* x   = (const float*)d_in[0];
  const float* mi  = (const float*)d_in[1];
  const float* lng = (const float*)d_in[2];
  const float* lnb = (const float*)d_in[3];
  const float* mrw = (const float*)d_in[4];
  const float* mrb = (const float*)d_in[5];
  const float* fw1 = (const float*)d_in[6];
  const float* fb1 = (const float*)d_in[7];
  const float* fw2 = (const float*)d_in[8];
  const float* fb2 = (const float*)d_in[9];
  float* out = (float*)d_out;

  // workspace carve-out (~218 MB)
  unsigned short* ws_fw1 = (unsigned short*)d_ws;            // 2048*4096 bf16
  unsigned short* ws_mrw = ws_fw1 + (size_t)H_DIM * HM_DIM;  // 16*2048 bf16
  unsigned short* ws_mi  = ws_mrw + (size_t)E_EXP * H_DIM;   // T*4096 bf16
  unsigned short* ws_xn  = ws_mi  + (size_t)T_TOK * HM_DIM;  // T*2048 bf16
  float* ws_mwacc = (float*)(ws_xn + (size_t)T_TOK * H_DIM); // T*2 (atomic)
  float* ws_pexp  = ws_mwacc + (size_t)T_TOK * 2;            // 16
  float* ws_msum  = ws_pexp + E_EXP;                         // 2
  float* ws_mw    = ws_msum + M_MOD;                         // T*2

  const int nzero = T_TOK * 2 + E_EXP + M_MOD;               // mwacc+pexp+msum
  zero_kernel<<<(nzero + 255) / 256, 256, 0, stream>>>(ws_mwacc, nzero);
  cvt_f32_bf16<<<4096, 256, 0, stream>>>(fw1, ws_fw1, H_DIM * HM_DIM);
  cvt_f32_bf16<<<32, 256, 0, stream>>>(mrw, ws_mrw, E_EXP * H_DIM);
  cvt_f32_bf16<<<8192, 256, 0, stream>>>(mi, ws_mi, T_TOK * HM_DIM);

  fusion_kernel<<<(T_TOK / MBLK) * (H_DIM / NBLK), FUS_THREADS, 0, stream>>>(
      ws_mi, ws_fw1, fb1, fw2, ws_mwacc);
  modweights_kernel<<<T_TOK / 256, 256, 0, stream>>>(ws_mwacc, fb2, ws_mw, ws_msum);
  ln_kernel<<<T_TOK, 256, 0, stream>>>(x, lng, lnb, ws_xn);
  router_kernel<<<T_TOK / 128, 256, 0, stream>>>(
      ws_xn, ws_mrw, mrb, ws_mw, out, ws_pexp);
  finalize_kernel<<<1, 32, 0, stream>>>(ws_pexp, ws_msum, out);
}
